// AttentionLayer_72258529788543
// MI455X (gfx1250) — matrix operand order repaired and checked by reading, not compiled, not run on hardware
//
#include <hip/hip_runtime.h>
#include <hip/hip_bf16.h>

typedef __attribute__((ext_vector_type(16))) _Float16 v16h;
typedef __attribute__((ext_vector_type(8)))  float    v8f;
typedef __attribute__((ext_vector_type(2)))  _Float16 h2;
typedef __attribute__((ext_vector_type(4)))  unsigned int u32x4;
typedef __attribute__((ext_vector_type(8)))  int i32x8;
typedef __attribute__((ext_vector_type(4)))  int i32x4;

#define Bb   8
#define S1   512
#define S2   512
#define Hd   128          // H1 == H2 == HID == 128
#define MROWS (Bb * S1)   // 4096

#if defined(__has_builtin)
#  if __has_builtin(__builtin_amdgcn_tensor_load_to_lds) && __has_builtin(__builtin_amdgcn_s_wait_tensorcnt)
#    define HAVE_TDM 1
#  endif
#endif
#ifndef HAVE_TDM
#  define HAVE_TDM 0
#endif

// ---------------------------------------------------------------------------
// Kernel 1: projections -> f16.  qp16 = query @ W1[:, :128]^T + b1,
//                                kp16 = key   @ W1[:, 128:]^T
// One wave computes one 16x16 tile; K chained over 4 x v_wmma_f32_16x16x32_f16.
// ---------------------------------------------------------------------------
__global__ __launch_bounds__(32) void proj_kernel(const float* __restrict__ query,
                                                  const float* __restrict__ key,
                                                  const float* __restrict__ W1,
                                                  const float* __restrict__ b1,
                                                  _Float16* __restrict__ qp16,
                                                  _Float16* __restrict__ kp16) {
    int tile  = blockIdx.x;
    int which = tile >> 11;            // 0 = qp (query), 1 = kp (key)
    tile &= 2047;
    int mt = tile >> 3;                // 0..255  (M tiles of 16)
    int nt = tile & 7;                 // 0..7    (N tiles of 16)

    int lane = threadIdx.x & 31;
    int g    = lane >> 4;
    int sr   = lane & 15;

    const float* X    = which ? key : query;
    const float* Wrow = W1 + (nt * 16 + sr) * 256 + (which ? 128 : 0);
    _Float16*    dst  = which ? kp16 : qp16;

    v8f acc = {};

    #pragma unroll
    for (int kb = 0; kb < 128; kb += 32) {
        // A fragment: row (mt*16+sr); K runs [kb+8g,+8) and [kb+16+8g,+8)
        const float* arow = X + (mt * 16 + sr) * Hd + kb + 8 * g;
        float av[16];
        *(float4*)&av[0]  = *(const float4*)(arow + 0);
        *(float4*)&av[4]  = *(const float4*)(arow + 4);
        *(float4*)&av[8]  = *(const float4*)(arow + 16);
        *(float4*)&av[12] = *(const float4*)(arow + 20);
        v16h af;
        #pragma unroll
        for (int i = 0; i < 16; ++i) af[i] = (_Float16)av[i];

        // B fragment: column n = nt*16+sr (a row of W1); K runs [kb+8g,+8) and [kb+16+8g,+8)
        const float* brow = Wrow + kb + 8 * g;
        float bv[16];
        *(float4*)&bv[0]  = *(const float4*)(brow + 0);
        *(float4*)&bv[4]  = *(const float4*)(brow + 4);
        *(float4*)&bv[8]  = *(const float4*)(brow + 16);
        *(float4*)&bv[12] = *(const float4*)(brow + 20);
        v16h bf;
        #pragma unroll
        for (int i = 0; i < 16; ++i) bf[i] = (_Float16)bv[i];

        acc = __builtin_amdgcn_wmma_f32_16x16x32_f16(false, af, false, bf,
                                                     (short)0, acc, false, false);
    }

    int   n    = nt * 16 + sr;
    int   m0   = mt * 16 + 8 * g;
    float bias = which ? 0.0f : b1[n];
    #pragma unroll
    for (int r = 0; r < 8; ++r)
        dst[(m0 + r) * Hd + n] = (_Float16)(acc[r] + bias);
}

// ---------------------------------------------------------------------------
// Kernel 2: value f32 [B,S2,H] -> f16 transposed [B][H][S2] for WMMA B-frags
// ---------------------------------------------------------------------------
__global__ void vconv_kernel(const float* __restrict__ value, _Float16* __restrict__ vt) {
    int idx = blockIdx.x * 256 + threadIdx.x;    // total B*S2*H = 524288
    if (idx >= Bb * S2 * Hd) return;
    int h = idx & 127;
    int t = (idx >> 7) & 511;
    int b = idx >> 16;
    vt[(b * Hd + h) * S2 + t] = (_Float16)value[idx];
}

// ---------------------------------------------------------------------------
// Kernel 3: fused score (packed f16) + mask + exp + P@V (WMMA) + normalize.
// One block (4 waves) per (batch, 16-row s tile).  kp is staged into LDS in
// 32KB chunks of 128 t-rows via the Tensor Data Mover; each wave then handles
// one 32-t sub-chunk per stage.
// ---------------------------------------------------------------------------
__global__ __launch_bounds__(128) void attn_kernel(const _Float16* __restrict__ qp16,
                                                   const _Float16* __restrict__ kp16,
                                                   const _Float16* __restrict__ vt,
                                                   const int* __restrict__ q_mask,
                                                   const int* __restrict__ k_mask,
                                                   const float* __restrict__ W2,
                                                   const float* __restrict__ b2,
                                                   float* __restrict__ out) {
    __shared__ _Float16 kpS[128 * Hd];      // 32 KB: current chunk of kp rows
    __shared__ _Float16 qpS[16 * 136];      // padded stride: 4-bank rotation/row
    __shared__ _Float16 w2S[128];
    __shared__ float    oS[16 * 128];
    __shared__ float    rsumS[16];
    __shared__ int      kmS[512];
    __shared__ int      qmS[16];

    int b     = blockIdx.x >> 5;
    int stile = blockIdx.x & 31;
    int tid   = threadIdx.x;
    int lane  = tid & 31;
    int wave  = tid >> 5;
    int g     = lane >> 4;
    int sr    = lane & 15;

    const _Float16* qpb = qp16 + ((size_t)(b * S1 + stile * 16)) * Hd;
    const _Float16* vtb = vt + (size_t)b * Hd * S2;

    for (int i = tid; i < 16 * 128; i += 128) {
        qpS[(i >> 7) * 136 + (i & 127)] = qpb[i];
        oS[i] = 0.0f;
    }
    for (int i = tid; i < 512; i += 128) kmS[i] = k_mask[b * S2 + i];
    if (tid < 128) w2S[tid] = (_Float16)W2[tid];
    if (tid < 16) { qmS[tid] = q_mask[b * S1 + stile * 16 + tid]; rsumS[tid] = 0.0f; }
    __syncthreads();

    float b2v = b2[0];
    int   qm  = qmS[sr];

    v8f acc[8];
    #pragma unroll
    for (int i = 0; i < 8; ++i) acc[i] = (v8f){};
    float rpart = 0.0f;

    const _Float16* qrow = &qpS[sr * 136];

    for (int mc = 0; mc < 4; ++mc) {           // 4 stages of 128 t-rows
        __syncthreads();                        // previous chunk fully consumed

#if HAVE_TDM
        if (wave == 0) {
            // ---- Tensor Data Mover: 2D tile (128 rows x 128 f16) -> LDS
            unsigned lds_off = (unsigned)(unsigned long long)(void*)kpS;
            unsigned long long ga =
                (unsigned long long)(const void*)(kp16 + ((size_t)b * S2 + (size_t)mc * 128) * Hd);
            // D# group0: count=1 | lds_addr | global_addr | type=2
            u32x4 g0 = { 1u, lds_off, (unsigned)ga,
                         (unsigned)((ga >> 32) & 0x01ffffffu) | 0x80000000u };
            // D# group1: data_size=2B; tensor_dim0=128, tensor_dim1=512;
            //            tile_dim0=128, tile_dim1=128; dim0_stride=128
            i32x8 g1 = { (int)(1u << 16),        // wg_mask=0, data_size=1 (2B)
                         (int)(128u << 16),      // tensor_dim0[15:0] @ bits 63:48
                         (int)(512u << 16),      // dim0 hi=0, tensor_dim1[15:0]
                         (int)(128u << 16),      // dim1 hi=0, tile_dim0=128
                         (int)128,               // tile_dim1=128, tile_dim2=0
                         (int)128,               // tensor_dim0_stride low32
                         0, 0 };                 // stride0 hi, stride1
            i32x4 z4 = { 0, 0, 0, 0 };
#if __clang_major__ >= 23
            i32x8 z8 = { 0, 0, 0, 0, 0, 0, 0, 0 };
            __builtin_amdgcn_tensor_load_to_lds(g0, g1, z4, z4, z8, 0);
#else
            __builtin_amdgcn_tensor_load_to_lds(g0, g1, z4, z4, 0);
#endif
            __builtin_amdgcn_s_wait_tensorcnt(0);
        }
        __syncthreads();
#else
        {   // fallback: cooperative vector copy
            const float4* src = (const float4*)(kp16 + ((size_t)b * S2 + (size_t)mc * 128) * Hd);
            float4* dstS = (float4*)kpS;
            for (int i = tid; i < (128 * Hd) / 8; i += 128) dstS[i] = src[i];
            __syncthreads();
        }
#endif

        int tloc  = wave * 32;                  // this wave's 32-t sub-chunk (local)
        int tbase = mc * 128 + tloc;            // global t base

        // ---- score in packed f16: pacc[j] = sum_h relu(qp+kp) * w2 (pairs)
        h2 pacc[16];
        #pragma unroll
        for (int j = 0; j < 16; ++j) pacc[j] = (h2){};

        for (int hb = 0; hb < Hd; hb += 16) {
            h2 q[8], w[8];
            *(float4*)&q[0] = *(const float4*)(qrow + hb);
            *(float4*)&q[4] = *(const float4*)(qrow + hb + 8);
            *(float4*)&w[0] = *(const float4*)(&w2S[hb]);
            *(float4*)&w[4] = *(const float4*)(&w2S[hb + 8]);
            #pragma unroll
            for (int j = 0; j < 16; ++j) {
                // j order == WMMA 16-bit A-fragment K layout for this lane
                int tl = tloc + 8 * g + j + ((j >= 8) ? 8 : 0);
                const _Float16* kr = &kpS[tl * Hd + hb];
                h2 k[8];
                *(float4*)&k[0] = *(const float4*)kr;
                *(float4*)&k[4] = *(const float4*)(kr + 8);
                h2 a = pacc[j];
                #pragma unroll
                for (int u = 0; u < 8; ++u) {
                    h2 r = __builtin_elementwise_max(q[u] + k[u], (h2){});
                    a = r * w[u] + a;           // v_pk_fma_f16
                }
                pacc[j] = a;
            }
        }

        // ---- mask + exp (== masked_fill(-inf).exp()), row-sum, pack A-frag
        v16h af;
        #pragma unroll
        for (int j = 0; j < 16; ++j) {
            int   t  = tbase + 8 * g + j + ((j >= 8) ? 8 : 0);
            float pj = (float)pacc[j].x + (float)pacc[j].y;
            float pv = (qm && kmS[t]) ? __expf(pj + b2v) : 0.0f;
            rpart += pv;
            af[j] = (_Float16)pv;
        }

        // ---- O += P @ V : 8 N-tiles of 16 columns, K = 32 t's per WMMA
        #pragma unroll
        for (int ht = 0; ht < 8; ++ht) {
            typedef __attribute__((ext_vector_type(8))) _Float16 v8h_;
            const _Float16* vb = vtb + (ht * 16 + sr) * S2 + tbase + 8 * g;
            v8h_ b0 = *(const v8h_*)vb;
            v8h_ b1v = *(const v8h_*)(vb + 16);
            v16h bf = __builtin_shufflevector(b0, b1v, 0,1,2,3,4,5,6,7,8,9,10,11,12,13,14,15);
            acc[ht] = __builtin_amdgcn_wmma_f32_16x16x32_f16(false, af, false, bf,
                                                             (short)0, acc[ht], false, false);
        }
    }

    // ---- combine wave partials (ds_add_f32)
    atomicAdd(&rsumS[sr], rpart);
    #pragma unroll
    for (int ht = 0; ht < 8; ++ht)
        #pragma unroll
        for (int r = 0; r < 8; ++r)
            atomicAdd(&oS[(r + 8 * g) * 128 + ht * 16 + sr], acc[ht][r]);
    __syncthreads();

    // ---- normalize: out = O / clip(rowsum, 2e-15)
    float* outb = out + (size_t)(b * S1 + stile * 16) * Hd;
    for (int i = tid; i < 16 * 128; i += 128) {
        int srow = i >> 7;
        outb[i] = oS[i] / fmaxf(rsumS[srow], 2e-15f);
    }
}

// ---------------------------------------------------------------------------
extern "C" void kernel_launch(void* const* d_in, const int* in_sizes, int n_in,
                              void* d_out, int out_size, void* d_ws, size_t ws_size,
                              hipStream_t stream) {
    const float* query = (const float*)d_in[0];
    const float* key   = (const float*)d_in[1];
    const float* value = (const float*)d_in[2];
    const int*   qmask = (const int*)d_in[3];
    const int*   kmask = (const int*)d_in[4];
    const float* W1    = (const float*)d_in[5];
    const float* b1    = (const float*)d_in[6];
    const float* W2    = (const float*)d_in[7];
    const float* b2    = (const float*)d_in[8];
    float*       out   = (float*)d_out;

    _Float16* qp16 = (_Float16*)d_ws;                       // [4096,128] f16 (1 MB)
    _Float16* kp16 = qp16 + (size_t)MROWS * Hd;             // [4096,128] f16 (1 MB)
    _Float16* vt   = kp16 + (size_t)MROWS * Hd;             // [B][H][S2] f16 (1 MB)

    proj_kernel<<<4096, 32, 0, stream>>>(query, key, W1, b1, qp16, kp16);
    vconv_kernel<<<(Bb * S2 * Hd + 255) / 256, 256, 0, stream>>>(value, vt);
    attn_kernel<<<Bb * (S1 / 16), 128, 0, stream>>>(qp16, kp16, vt, qmask, kmask, W2, b2, out);
}
